// GroupQueryAttention_70884140253646
// MI455X (gfx1250) — matrix operand order repaired and checked by reading, not compiled, not run on hardware
//
#include <hip/hip_runtime.h>
#include <hip/hip_bf16.h>
#include <type_traits>

typedef __attribute__((ext_vector_type(16))) __bf16 v16bf;
typedef __attribute__((ext_vector_type(8)))  __bf16 v8bf;
typedef __attribute__((ext_vector_type(4)))  __bf16 v4bf;
typedef __attribute__((ext_vector_type(8)))  float  v8f;
typedef __attribute__((ext_vector_type(4)))  float  v4f;
typedef __attribute__((ext_vector_type(4)))  unsigned int u32x4;
typedef __attribute__((ext_vector_type(8)))  int          i32x8;
typedef __attribute__((ext_vector_type(4)))  int          i32x4;

#if defined(__HIP_DEVICE_COMPILE__) && \
    __has_builtin(__builtin_amdgcn_tensor_load_to_lds) && \
    __has_builtin(__builtin_amdgcn_s_wait_tensorcnt)
#define USE_TDM 1
#else
#define USE_TDM 0
#endif

// ---------------------------------------------------------------------------
// WMMA helper: D = A(16x32 bf16) * B(32x16 bf16) + C(16x16 f32)
// ---------------------------------------------------------------------------
__device__ __forceinline__ v8f wmma_bf16(v16bf a, v16bf b, v8f c) {
  return __builtin_amdgcn_wmma_f32_16x16x32_bf16(
      /*neg_a=*/false, a, /*neg_b=*/false, b,
      /*c_mod=*/(short)0, c, /*reuse_a=*/false, /*reuse_b=*/false);
}

// A-matrix fragment (16x32 bf16) from row-major LDS tile [16 rows][ld].
// ISA: lanes 0-15 -> M=lane, K={0..7,16..23}; lanes 16-31 -> K={8..15,24..31}.
__device__ __forceinline__ v16bf frag_a(const __bf16* tile, int ld, int lane) {
  const int half = lane >> 4;
  const int r = lane & 15;
  const __bf16* p = tile + r * ld + half * 8;
  v8bf lo = *(const v8bf*)(p);
  v8bf hi = *(const v8bf*)(p + 16);
  v16bf out;
#pragma unroll
  for (int i = 0; i < 8; ++i) { out[i] = lo[i]; out[i + 8] = hi[i]; }
  return out;
}

// B-matrix fragment (32x16 bf16) from K-contiguous LDS tile Bt[16 cols][ld].
// ISA: lane = column N=lane&15; lanes 0-15 hold K={0..7,16..23}, lanes 16-31 K={8..15,24..31}.
__device__ __forceinline__ v16bf frag_b(const __bf16* tile, int ld, int lane) {
  const int half = lane >> 4;
  const int c = lane & 15;
  const __bf16* p = tile + c * ld + half * 8;
  v8bf lo = *(const v8bf*)(p);
  v8bf hi = *(const v8bf*)(p + 16);
  v16bf out;
#pragma unroll
  for (int i = 0; i < 8; ++i) { out[i] = lo[i]; out[i + 8] = hi[i]; }
  return out;
}

#if USE_TDM
// LDS byte offset of a shared-memory pointer (generic -> AS3 addrspacecast).
__device__ __forceinline__ unsigned int lds_off(const void* p) {
  return (unsigned int)(uintptr_t)(__attribute__((address_space(3))) const void*)p;
}

// TDM 2D bf16 tile load: tile_d1 rows x tile_d0 elements (2B each) from a
// row-major tensor with dim0 stride stride_d0 (elements), into LDS with
// pad_enable inserting (pa+1) DWORDs after every 2^(pi+1) DWORDs stored.
// D# packing per CDNA5 ISA ch.8 (group0: count/lds_addr/global_addr/type=2;
// group1: data_size=2B, pads, dims, tile dims, dim0 stride; groups2/3: benign
// 3D/4D/5D extents of 1 so higher dims never trip OOB).
__device__ __forceinline__ void tdm_load_2d(
    unsigned int lds_byte_addr, const void* gaddr,
    unsigned int tile_d0, unsigned int tile_d1,
    unsigned int tensor_d0, unsigned int tensor_d1,
    unsigned long long stride_d0,
    unsigned int pi, unsigned int pa) {
  const unsigned long long ga = (unsigned long long)(uintptr_t)gaddr;
  u32x4 g0;
  g0[0] = 1u;                                    // count=1 (valid user D#)
  g0[1] = lds_byte_addr;                         // lds_addr
  g0[2] = (unsigned int)ga;                      // global_addr[31:0]
  g0[3] = ((unsigned int)(ga >> 32) & 0x01FFFFFFu) | (2u << 30);  // addr[56:32]|type=2
  i32x8 g1;
  g1[0] = (int)((1u << 16)                       // data_size = 2 bytes
              | (1u << 20)                       // pad_enable
              | (pi << 22) | (pa << 25));        // pad_interval / pad_amount
  g1[1] = (int)((tensor_d0 & 0xFFFFu) << 16);    // atomic_addr=0 | dim0[15:0]
  g1[2] = (int)(((tensor_d0 >> 16) & 0xFFFFu) | ((tensor_d1 & 0xFFFFu) << 16));
  g1[3] = (int)(((tensor_d1 >> 16) & 0xFFFFu) | (tile_d0 << 16));
  g1[4] = (int)(tile_d1 & 0xFFFFu);              // tile_dim1 | tile_dim2=0
  g1[5] = (int)(unsigned int)(stride_d0 & 0xFFFFFFFFu);
  g1[6] = (int)(unsigned int)((stride_d0 >> 32) & 0xFFFFu);  // | dim1_stride lo=0
  g1[7] = 0;
  i32x4 g2;
  g2[0] = 1;                                     // tensor_dim2 = 1
  g2[1] = 1;                                     // tensor_dim3 = 1
  g2[2] = 0; g2[3] = 0;                          // dim2 stride / tile_dim3 = 0
  i32x4 g3;
  g3[0] = 0;                                     // dim3 stride lo
  g3[1] = (int)(1u << 16);                       // dim3 stride hi=0 | tensor_dim4=1
  g3[2] = 0; g3[3] = 0;                          // tile_dim4 = 0
#if defined(__clang_major__) && (__clang_major__ >= 23)
  i32x8 gz;
#pragma unroll
  for (int i = 0; i < 8; ++i) gz[i] = 0;
  __builtin_amdgcn_tensor_load_to_lds(g0, g1, g2, g3, gz, 0);
#else
  __builtin_amdgcn_tensor_load_to_lds(g0, g1, g2, g3, 0);
#endif
}
#endif  // USE_TDM

// ---------------------------------------------------------------------------
// Weight transpose + fp32 -> bf16 convert:  W[K][N] -> Wt[N][K]
// ---------------------------------------------------------------------------
__global__ __launch_bounds__(256) void transpose_cvt(
    const float* __restrict__ W, __bf16* __restrict__ Wt, int K, int N) {
  size_t idx = (size_t)blockIdx.x * 256 + threadIdx.x;
  size_t total = (size_t)K * N;
  if (idx >= total) return;
  int k = (int)(idx % K);
  size_t n = idx / K;
  Wt[idx] = (__bf16)W[(size_t)k * N + n];
}

// ---------------------------------------------------------------------------
// GEMM: C[M][N] = A[M][K] @ Wt[N][K]^T + bias.  BM=BN=128, BK=64.
// 256 threads = 8 waves in a 4(m) x 2(n) grid; each wave owns 32x64 (2x4 WMMA,
// 16 wmma per K-step).  B tile staged by TDM when available.
// ---------------------------------------------------------------------------
template <typename AT, typename OT>
__global__ __launch_bounds__(256) void gemm_bias_wmma(
    const AT* __restrict__ A, const __bf16* __restrict__ Bt,
    const float* __restrict__ bias, OT* __restrict__ C,
    int M, int N, int K) {
  constexpr int LDT = 72;  // 144B row stride: 16B-aligned, spreads LDS banks
  __shared__ __bf16 As[128 * LDT];
  __shared__ __bf16 Bs[128 * LDT];

  const int t = threadIdx.x;
  const int lane = t & 31;
  const int w = t >> 5;
  const int wm = w >> 1;  // 0..3
  const int wn = w & 1;   // 0..1
  const int bm = blockIdx.y * 128;
  const int bn = blockIdx.x * 128;

  v8f acc[2][4];
#pragma unroll
  for (int i = 0; i < 2; ++i)
#pragma unroll
    for (int j = 0; j < 4; ++j)
#pragma unroll
      for (int v = 0; v < 8; ++v) acc[i][j][v] = 0.f;

  for (int k0 = 0; k0 < K; k0 += 64) {
    __syncthreads();
#if USE_TDM
    if (t == 0) {
      // 128 rows x 64 elems (128B) per row; pad 16B/row -> 144B LDS stride
      tdm_load_2d(lds_off(Bs), Bt + (size_t)bn * K + k0,
                  /*tile_d0=*/64, /*tile_d1=*/128,
                  /*tensor_d0=*/(unsigned)K, /*tensor_d1=*/(unsigned)N,
                  /*stride_d0=*/(unsigned long long)K, /*pi=*/4, /*pa=*/3);
    }
#endif
    // ---- stage A tile (128 x 64) ----
    if constexpr (std::is_same<AT, float>::value) {
      const int r0 = t >> 4;          // 0..15
      const int cc = (t & 15) * 4;    // 0..60
#pragma unroll
      for (int p = 0; p < 8; ++p) {
        const int row = r0 + p * 16;
        const float* gp = A + (size_t)(bm + row) * K + k0 + cc;
        v4f g = *(const v4f*)gp;
        __builtin_prefetch(gp + 64, 0, 1);
        v4bf cv;
#pragma unroll
        for (int i = 0; i < 4; ++i) cv[i] = (__bf16)g[i];
        *(v4bf*)(As + row * LDT + cc) = cv;
      }
    } else {
      const int r0 = t >> 3;          // 0..31
      const int cc = (t & 7) * 8;     // 0..56
#pragma unroll
      for (int p = 0; p < 4; ++p) {
        const int row = r0 + p * 32;
        const __bf16* gp = A + (size_t)(bm + row) * K + k0 + cc;
        *(v8bf*)(As + row * LDT + cc) = *(const v8bf*)gp;
        __builtin_prefetch(gp + 64, 0, 1);
      }
    }
#if USE_TDM
    __builtin_amdgcn_s_wait_tensorcnt(0);
#else
    // ---- stage B tile (128 x 64, already K-contiguous bf16) ----
    {
      const int r0 = t >> 3;
      const int cc = (t & 7) * 8;
#pragma unroll
      for (int p = 0; p < 4; ++p) {
        const int row = r0 + p * 32;
        *(v8bf*)(Bs + row * LDT + cc) =
            *(const v8bf*)(Bt + (size_t)(bn + row) * K + k0 + cc);
      }
    }
#endif
    __syncthreads();

#pragma unroll
    for (int dk = 0; dk < 2; ++dk) {
      v16bf fa[2];
#pragma unroll
      for (int mt = 0; mt < 2; ++mt)
        fa[mt] = frag_a(As + (wm * 32 + mt * 16) * LDT + dk * 32, LDT, lane);
#pragma unroll
      for (int nt = 0; nt < 4; ++nt) {
        v16bf fb = frag_b(Bs + (wn * 64 + nt * 16) * LDT + dk * 32, LDT, lane);
#pragma unroll
        for (int mt = 0; mt < 2; ++mt)
          acc[mt][nt] = wmma_bf16(fa[mt], fb, acc[mt][nt]);
      }
    }
  }

  // Epilogue: C layout -> row = v + (lane>=16)*8, col = lane&15 within 16x16 tile
  const int half = lane >> 4;
  const int cl = lane & 15;
#pragma unroll
  for (int mt = 0; mt < 2; ++mt)
#pragma unroll
    for (int nt = 0; nt < 4; ++nt) {
      const int col = bn + wn * 64 + nt * 16 + cl;
      const float bvv = bias[col];
#pragma unroll
      for (int v = 0; v < 8; ++v) {
        const int row = bm + wm * 32 + mt * 16 + v + half * 8;
        C[(size_t)row * N + col] = (OT)(acc[mt][nt][v] + bvv);
      }
    }
}

// ---------------------------------------------------------------------------
// Flash GQA attention.  Grid: (S/128, NUM_Q=32, B).  256 threads = 8 waves,
// each wave owns 16 query rows.  Iterates 16 key blocks of 128.
// Q/K tiles staged by TDM (row = 256B + 16B pad -> 272B LDS stride).
// ---------------------------------------------------------------------------
__global__ __launch_bounds__(256) void gqa_flash_wmma(
    const __bf16* __restrict__ Qp, const __bf16* __restrict__ Kp,
    const __bf16* __restrict__ Vp, __bf16* __restrict__ Yp) {
  constexpr int LDT = 136;  // 272B stride, 16B-aligned, bank-spreading pad
  constexpr int S = 2048, E = 4096, KVD = 1024, D = 128;
  __shared__ __bf16 KsS[128 * LDT];      // K block rows=keys (also Q staging)
  __shared__ __bf16 VtS[128 * LDT];      // V block transposed: [d][key]
  __shared__ __bf16 PsS[8 * 16 * LDT];   // per-wave P tiles

  const int t = threadIdx.x;
  const int lane = t & 31;
  const int w = t >> 5;
  const int half = lane >> 4;
  const int cl = lane & 15;
  const int qb = blockIdx.x;
  const int hq = blockIdx.y;
  const int b = blockIdx.z;
  const int kv = hq >> 2;  // GROUPS = 4
  const size_t qrow0 = (size_t)b * S + (size_t)qb * 128;

  // ---- stage Q (128 x 128) into KsS, build register A-fragments ----
#if USE_TDM
  if (t == 0)
    tdm_load_2d(lds_off(KsS), Qp + qrow0 * E + hq * D,
                /*tile_d0=*/128, /*tile_d1=*/128,
                /*tensor_d0=*/E, /*tensor_d1=*/2 * S,
                /*stride_d0=*/(unsigned long long)E, /*pi=*/5, /*pa=*/3);
  __builtin_amdgcn_s_wait_tensorcnt(0);
#else
  {
    const int row = t >> 1;
    const int c0 = (t & 1) * 64;
    const __bf16* g = Qp + (qrow0 + row) * E + hq * D + c0;
#pragma unroll
    for (int j = 0; j < 8; ++j)
      *(v8bf*)(KsS + row * LDT + c0 + j * 8) = *(const v8bf*)(g + j * 8);
  }
#endif
  __syncthreads();
  v16bf fq[4];
#pragma unroll
  for (int dk = 0; dk < 4; ++dk)
    fq[dk] = frag_a(KsS + (w * 16) * LDT + dk * 32, LDT, lane);

  v8f acco[8];
#pragma unroll
  for (int i = 0; i < 8; ++i)
#pragma unroll
    for (int v = 0; v < 8; ++v) acco[i][v] = 0.f;
  float mrow[8], lrow[8];
#pragma unroll
  for (int v = 0; v < 8; ++v) { mrow[v] = -1.0e30f; lrow[v] = 0.f; }
  const float SC = 1.44269504089f * 0.088388347648f;  // log2(e)/sqrt(128)
  __bf16* Pw = PsS + w * 16 * LDT;

  for (int kb = 0; kb < 16; ++kb) {
    const size_t krow0 = (size_t)b * S + (size_t)kb * 128;
    __syncthreads();  // all waves done reading KsS/VtS (or Q frags built)
#if USE_TDM
    if (t == 0)
      tdm_load_2d(lds_off(KsS), Kp + krow0 * KVD + kv * D,
                  /*tile_d0=*/128, /*tile_d1=*/128,
                  /*tensor_d0=*/KVD, /*tensor_d1=*/2 * S,
                  /*stride_d0=*/(unsigned long long)KVD, /*pi=*/5, /*pa=*/3);
#endif
    {
      const int row = t >> 1;
      const int c0 = (t & 1) * 64;
#if !USE_TDM
      const __bf16* gk = Kp + (krow0 + row) * KVD + kv * D + c0;
#pragma unroll
      for (int j = 0; j < 8; ++j)
        *(v8bf*)(KsS + row * LDT + c0 + j * 8) = *(const v8bf*)(gk + j * 8);
#endif
      const __bf16* gv = Vp + (krow0 + row) * KVD + kv * D + c0;
#pragma unroll
      for (int j = 0; j < 8; ++j) {
        v8bf vv = *(const v8bf*)(gv + j * 8);
#pragma unroll
        for (int e = 0; e < 8; ++e) VtS[(c0 + j * 8 + e) * LDT + row] = vv[e];
      }
    }
#if USE_TDM
    __builtin_amdgcn_s_wait_tensorcnt(0);
#endif
    __syncthreads();

    // ---- scores: S = Q @ K^T  (8 n-tiles x 4 k-chunks of 32) ----
    v8f s[8];
#pragma unroll
    for (int jn = 0; jn < 8; ++jn)
#pragma unroll
      for (int v = 0; v < 8; ++v) s[jn][v] = 0.f;
#pragma unroll
    for (int jn = 0; jn < 8; ++jn)
#pragma unroll
      for (int dk = 0; dk < 4; ++dk) {
        v16bf fb = frag_b(KsS + (jn * 16) * LDT + dk * 32, LDT, lane);
        s[jn] = wmma_bf16(fq[dk], fb, s[jn]);
      }
#pragma unroll
    for (int jn = 0; jn < 8; ++jn)
#pragma unroll
      for (int v = 0; v < 8; ++v) s[jn][v] *= SC;

    // ---- online softmax (per row v + half*8; cols live across 16 lanes) ----
    float alpha[8];
#pragma unroll
    for (int v = 0; v < 8; ++v) {
      float mx = s[0][v];
#pragma unroll
      for (int jn = 1; jn < 8; ++jn) mx = fmaxf(mx, s[jn][v]);
#pragma unroll
      for (int off = 1; off < 16; off <<= 1)
        mx = fmaxf(mx, __shfl_xor(mx, off, 32));
      const float mo = mrow[v];
      const float mn = fmaxf(mo, mx);
      mrow[v] = mn;
      const float al = __builtin_exp2f(mo - mn);
      alpha[v] = al;
      float rs = 0.f;
#pragma unroll
      for (int jn = 0; jn < 8; ++jn) {
        const float p = __builtin_exp2f(s[jn][v] - mn);
        s[jn][v] = p;
        rs += p;
      }
#pragma unroll
      for (int off = 1; off < 16; off <<= 1) rs += __shfl_xor(rs, off, 32);
      lrow[v] = lrow[v] * al + rs;
    }
#pragma unroll
    for (int jd = 0; jd < 8; ++jd)
#pragma unroll
      for (int v = 0; v < 8; ++v) acco[jd][v] *= alpha[v];

    // ---- P (C layout) -> LDS -> A layout; same-wave cross-lane: wait DS ----
#pragma unroll
    for (int jn = 0; jn < 8; ++jn)
#pragma unroll
      for (int v = 0; v < 8; ++v)
        Pw[(v + half * 8) * LDT + jn * 16 + cl] = (__bf16)s[jn][v];
    asm volatile("s_wait_dscnt 0" ::: "memory");

    v16bf fp[4];
#pragma unroll
    for (int kc = 0; kc < 4; ++kc) fp[kc] = frag_a(Pw + kc * 32, LDT, lane);

    // ---- O += P @ V  (8 d-tiles x 4 key-chunks of 32) ----
#pragma unroll
    for (int jd = 0; jd < 8; ++jd)
#pragma unroll
      for (int kc = 0; kc < 4; ++kc) {
        v16bf fb = frag_b(VtS + (jd * 16) * LDT + kc * 32, LDT, lane);
        acco[jd] = wmma_bf16(fp[kc], fb, acco[jd]);
      }
  }

  // ---- normalize and store y (bf16) ----
#pragma unroll
  for (int v = 0; v < 8; ++v) {
    const float inv = 1.0f / lrow[v];
    const size_t row = qrow0 + (size_t)w * 16 + v + half * 8;
#pragma unroll
    for (int jd = 0; jd < 8; ++jd) {
      const int col = hq * D + jd * 16 + cl;
      Yp[row * E + col] = (__bf16)(acco[jd][v] * inv);
    }
  }
}

// ---------------------------------------------------------------------------
// Host-side launcher
// ---------------------------------------------------------------------------
extern "C" void kernel_launch(void* const* d_in, const int* in_sizes, int n_in,
                              void* d_out, int out_size, void* d_ws, size_t ws_size,
                              hipStream_t stream) {
  (void)in_sizes; (void)n_in; (void)out_size; (void)ws_size;
  const float* query = (const float*)d_in[0];
  const float* keyi  = (const float*)d_in[1];
  const float* value = (const float*)d_in[2];
  const float* Wq    = (const float*)d_in[3];
  const float* bq    = (const float*)d_in[4];
  const float* Wk    = (const float*)d_in[5];
  const float* bk    = (const float*)d_in[6];
  const float* Wv    = (const float*)d_in[7];
  const float* bv    = (const float*)d_in[8];
  const float* Wo    = (const float*)d_in[9];
  const float* bo    = (const float*)d_in[10];
  float* out = (float*)d_out;

  const int B = 2, S = 2048, E = 4096, KVD = 1024, M = B * S;

  __bf16* ws = (__bf16*)d_ws;
  size_t off = 0;
  __bf16* WqT = ws + off; off += (size_t)E * E;
  __bf16* WkT = ws + off; off += (size_t)E * KVD;
  __bf16* WvT = ws + off; off += (size_t)E * KVD;
  __bf16* WoT = ws + off; off += (size_t)E * E;
  __bf16* Qp  = ws + off; off += (size_t)M * E;
  __bf16* Kpj = ws + off; off += (size_t)M * KVD;
  __bf16* Vpj = ws + off; off += (size_t)M * KVD;
  __bf16* Yp  = ws + off; off += (size_t)M * E;

  dim3 blk(256);
  // one-time weight convert+transpose (W[K][N] fp32 -> Wt[N][K] bf16)
  transpose_cvt<<<(unsigned)(((size_t)E * E) / 256), blk, 0, stream>>>(Wq, WqT, E, E);
  transpose_cvt<<<(unsigned)(((size_t)E * KVD) / 256), blk, 0, stream>>>(Wk, WkT, E, KVD);
  transpose_cvt<<<(unsigned)(((size_t)E * KVD) / 256), blk, 0, stream>>>(Wv, WvT, E, KVD);
  transpose_cvt<<<(unsigned)(((size_t)E * E) / 256), blk, 0, stream>>>(Wo, WoT, E, E);

  // projections (fp32 activations -> bf16 Q/K/V)
  gemm_bias_wmma<float, __bf16><<<dim3(E / 128, M / 128), blk, 0, stream>>>(
      query, WqT, bq, Qp, M, E, E);
  gemm_bias_wmma<float, __bf16><<<dim3(KVD / 128, M / 128), blk, 0, stream>>>(
      keyi, WkT, bk, Kpj, M, KVD, E);
  gemm_bias_wmma<float, __bf16><<<dim3(KVD / 128, M / 128), blk, 0, stream>>>(
      value, WvT, bv, Vpj, M, KVD, E);

  // flash GQA attention
  gqa_flash_wmma<<<dim3(S / 128, 32, B), blk, 0, stream>>>(Qp, Kpj, Vpj, Yp);

  // output projection (bf16 -> fp32 out)
  gemm_bias_wmma<__bf16, float><<<dim3(E / 128, M / 128), blk, 0, stream>>>(
      Yp, WoT, bo, out, M, E, E);
}
